// SwitchBackGlobalMLP_6863357739204
// MI455X (gfx1250) — 8-bit matrix operand order repaired and checked by reading, compiled, not run on hardware
//
#include <hip/hip_runtime.h>
#include <hip/hip_bf16.h>


typedef int v8i __attribute__((ext_vector_type(8)));

#define LDS_ROW 80

__device__ __forceinline__ float clamp127(float v) {
    return fminf(127.0f, fmaxf(-127.0f, rintf(v)));
}

__device__ __forceinline__ float gelu_tanh(float x) {
    const float k0 = 0.7978845608028654f;
    const float k1 = 0.044715f;
    float x3 = x * x * x;
    return 0.5f * x * (1.0f + tanhf(k0 * (x + k1 * x3)));
}

__device__ __forceinline__ void async_b128(unsigned lds_off, const signed char* g) {
    asm volatile("global_load_async_to_lds_b128 %0, %1, off"
                 :: "v"(lds_off), "v"(g)
                 : "memory");
}

__global__ __launch_bounds__(256) void init_zero_kernel(unsigned* wbits, int nw,
                                                        unsigned* rowbits, int nr) {
    int i = blockIdx.x * blockDim.x + threadIdx.x;
    int stride = gridDim.x * blockDim.x;
    if (i < nw) wbits[i] = 0u;
    for (int j = i; j < nr; j += stride) rowbits[j] = 0u;
}

__global__ __launch_bounds__(256) void absmax_reduce_kernel(const float* __restrict__ X,
                                                            unsigned* __restrict__ outBits,
                                                            size_t n4) {
    const float4* X4 = reinterpret_cast<const float4*>(X);
    float m = 0.0f;
    size_t stride = (size_t)gridDim.x * blockDim.x;
    for (size_t i = (size_t)blockIdx.x * blockDim.x + threadIdx.x; i < n4; i += stride) {
        float4 v = X4[i];
        m = fmaxf(m, fmaxf(fmaxf(fabsf(v.x), fabsf(v.y)), fmaxf(fabsf(v.z), fabsf(v.w))));
    }
#pragma unroll
    for (int off = 16; off; off >>= 1) m = fmaxf(m, __shfl_xor(m, off, 32));
    __shared__ float red[8];
    if ((threadIdx.x & 31) == 0) red[threadIdx.x >> 5] = m;
    __syncthreads();
    if (threadIdx.x == 0) {
        float t = red[0];
#pragma unroll
        for (int i = 1; i < 8; ++i) t = fmaxf(t, red[i]);
        atomicMax(outBits, __float_as_uint(t));
    }
}

__global__ __launch_bounds__(256) void quant_global_kernel(const float* __restrict__ W,
                                                           signed char* __restrict__ Wq,
                                                           const unsigned* __restrict__ bits,
                                                           size_t n4) {
    const float inv = 127.0f / __uint_as_float(*bits);
    const float4* W4 = reinterpret_cast<const float4*>(W);
    char4* Q4 = reinterpret_cast<char4*>(Wq);
    size_t stride = (size_t)gridDim.x * blockDim.x;
    for (size_t i = (size_t)blockIdx.x * blockDim.x + threadIdx.x; i < n4; i += stride) {
        float4 v = W4[i];
        char4 q;
        q.x = (signed char)clamp127(v.x * inv);
        q.y = (signed char)clamp127(v.y * inv);
        q.z = (signed char)clamp127(v.z * inv);
        q.w = (signed char)clamp127(v.w * inv);
        Q4[i] = q;
    }
}

__global__ __launch_bounds__(256) void quant_rows_kernel(const float* __restrict__ X,
                                                         signed char* __restrict__ Xq,
                                                         float* __restrict__ sOut, int K) {
    const int row = blockIdx.x;
    const float4* xr = reinterpret_cast<const float4*>(X + (size_t)row * K);
    char4* qr = reinterpret_cast<char4*>(Xq + (size_t)row * K);
    const int K4 = K >> 2;
    float m = 0.0f;
    for (int i = threadIdx.x; i < K4; i += blockDim.x) {
        float4 v = xr[i];
        m = fmaxf(m, fmaxf(fmaxf(fabsf(v.x), fabsf(v.y)), fmaxf(fabsf(v.z), fabsf(v.w))));
    }
#pragma unroll
    for (int off = 16; off; off >>= 1) m = fmaxf(m, __shfl_xor(m, off, 32));
    __shared__ float red[8];
    __shared__ float rowmax;
    if ((threadIdx.x & 31) == 0) red[threadIdx.x >> 5] = m;
    __syncthreads();
    if (threadIdx.x == 0) {
        float t = red[0];
#pragma unroll
        for (int i = 1; i < 8; ++i) t = fmaxf(t, red[i]);
        rowmax = t;
        sOut[row] = t;
    }
    __syncthreads();
    const float inv = 127.0f / rowmax;
    for (int i = threadIdx.x; i < K4; i += blockDim.x) {
        float4 v = xr[i];
        char4 q;
        q.x = (signed char)clamp127(v.x * inv);
        q.y = (signed char)clamp127(v.y * inv);
        q.z = (signed char)clamp127(v.z * inv);
        q.w = (signed char)clamp127(v.w * inv);
        qr[i] = q;
    }
}

__global__ __launch_bounds__(256) void quant_rows_from_bits_kernel(const float* __restrict__ X,
                                                                   const unsigned* __restrict__ rowBits,
                                                                   signed char* __restrict__ Xq,
                                                                   int K, size_t n4) {
    const float4* X4 = reinterpret_cast<const float4*>(X);
    char4* Q4 = reinterpret_cast<char4*>(Xq);
    const int K4 = K >> 2;
    size_t stride = (size_t)gridDim.x * blockDim.x;
    for (size_t i = (size_t)blockIdx.x * blockDim.x + threadIdx.x; i < n4; i += stride) {
        int row = (int)(i / (size_t)K4);
        const float inv = 127.0f / __uint_as_float(rowBits[row]);
        float4 v = X4[i];
        char4 q;
        q.x = (signed char)clamp127(v.x * inv);
        q.y = (signed char)clamp127(v.y * inv);
        q.z = (signed char)clamp127(v.z * inv);
        q.w = (signed char)clamp127(v.w * inv);
        Q4[i] = q;
    }
}

__device__ __forceinline__ v8i lds_a_frag(const signed char* tile, int m0, int lane) {
    const signed char* p = tile + (m0 + (lane & 15)) * LDS_ROW + ((lane >> 4) << 4);
    v8i a;
#pragma unroll
    for (int c = 0; c < 4; ++c) {
        int2 d = *reinterpret_cast<const int2*>(p + (c >> 1) * 32 + (c & 1) * 8);
        a[2 * c]     = d.x;
        a[2 * c + 1] = d.y;
    }
    return a;
}

__device__ __forceinline__ v8i lds_b_frag(const signed char* tile, int n0, int lane) {
    const signed char* p = tile + (n0 + (lane & 15)) * LDS_ROW + ((lane >> 4) << 4);
    int4 d0 = *reinterpret_cast<const int4*>(p);
    int4 d1 = *reinterpret_cast<const int4*>(p + 32);
    v8i b;
    b[0] = d0.x; b[1] = d0.y; b[2] = d0.z; b[3] = d0.w;
    b[4] = d1.x; b[5] = d1.y; b[6] = d1.z; b[7] = d1.w;
    return b;
}

template <bool GELU>
__global__ __launch_bounds__(256, 1) void gemm_iu8_kernel(
    const signed char* __restrict__ Aq, const signed char* __restrict__ Bq,
    const float* __restrict__ sArow, const unsigned* __restrict__ sWbits,
    const float* __restrict__ bias, float* __restrict__ Out,
    unsigned* __restrict__ rowBits, int M, int N, int K) {
    __shared__ __align__(16) signed char ldsA[2][256 * LDS_ROW];
    __shared__ __align__(16) signed char ldsB[2][128 * LDS_ROW];

    const int tid = threadIdx.x;
    const int lane = tid & 31;
    const int wave = tid >> 5;
    const int waveM = wave >> 1;
    const int waveN = wave & 1;
    const int mBlk = blockIdx.y * 256;
    const int nBlk = blockIdx.x * 128;
    const int mW = waveM * 64;
    const int nW = waveN * 64;

    const int rA[4] = {(tid + 0) >> 2, (tid + 256) >> 2, (tid + 512) >> 2, (tid + 768) >> 2};
    const int cA = (tid & 3) << 4;
    const int rB[2] = {(tid + 0) >> 2, (tid + 256) >> 2};

    auto issue_stage = [&](int buf, int k0) {
        unsigned aBase = (unsigned)(size_t)&ldsA[buf][0];
        unsigned bBase = (unsigned)(size_t)&ldsB[buf][0];
#pragma unroll
        for (int q = 0; q < 4; ++q)
            async_b128(aBase + (unsigned)(rA[q] * LDS_ROW + cA),
                       Aq + (size_t)(mBlk + rA[q]) * K + k0 + cA);
#pragma unroll
        for (int q = 0; q < 2; ++q)
            async_b128(bBase + (unsigned)(rB[q] * LDS_ROW + cA),
                       Bq + (size_t)(nBlk + rB[q]) * K + k0 + cA);
    };

    v8i acc[4][4];
#pragma unroll
    for (int i = 0; i < 4; ++i)
#pragma unroll
        for (int j = 0; j < 4; ++j) acc[i][j] = (v8i){0, 0, 0, 0, 0, 0, 0, 0};

    const int T = K >> 6;
    issue_stage(0, 0);
    for (int t = 0; t < T; ++t) {
        const int cur = t & 1;
        if (t + 1 < T) {
            issue_stage(cur ^ 1, (t + 1) << 6);
            asm volatile("s_wait_asynccnt 6" ::: "memory");
        } else {
            asm volatile("s_wait_asynccnt 0" ::: "memory");
        }
        __syncthreads();

        const signed char* bufA = &ldsA[cur][0];
        const signed char* bufB = &ldsB[cur][0];
        v8i bf[4];
#pragma unroll
        for (int j = 0; j < 4; ++j) bf[j] = lds_b_frag(bufB, nW + j * 16, lane);
#pragma unroll
        for (int i = 0; i < 4; ++i) {
            v8i af = lds_a_frag(bufA, mW + i * 16, lane);
#pragma unroll
            for (int j = 0; j < 4; ++j)
                acc[i][j] = __builtin_amdgcn_wmma_i32_16x16x64_iu8(
                    true, af, true, bf[j], acc[i][j], false, false);
        }
        __syncthreads();
    }

    const float sWf = __uint_as_float(*sWbits) * (1.0f / 16129.0f);
    const int m0 = mBlk + mW;
    const int n0 = nBlk + nW;
    const int colLo = lane & 15;
    float bv[4];
#pragma unroll
    for (int j = 0; j < 4; ++j) bv[j] = bias[n0 + j * 16 + colLo];

#pragma unroll
    for (int i = 0; i < 4; ++i) {
        const int rowBase = m0 + i * 16 + ((lane >> 4) << 3);
#pragma unroll
        for (int r = 0; r < 8; ++r) {
            const int row = rowBase + r;
            const float sc = sArow[row] * sWf;
            float am = 0.0f;
#pragma unroll
            for (int j = 0; j < 4; ++j) {
                const int col = n0 + j * 16 + colLo;
                float x = (float)acc[i][j][r] * sc + bv[j];
                if (GELU) {
                    x = gelu_tanh(x);
                    am = fmaxf(am, fabsf(x));
                }
                Out[(size_t)row * N + col] = x;
            }
            if (GELU) {
#pragma unroll
                for (int off = 8; off; off >>= 1) am = fmaxf(am, __shfl_xor(am, off, 32));
                if (colLo == 0) atomicMax(rowBits + row, __float_as_uint(am));
            }
        }
    }
}

extern "C" void kernel_launch(void* const* d_in, const int* in_sizes, int n_in,
                              void* d_out, int out_size, void* d_ws, size_t ws_size,
                              hipStream_t stream) {
    const float* x  = (const float*)d_in[0];
    const float* W1 = (const float*)d_in[1];
    const float* B1 = (const float*)d_in[2];
    const float* W2 = (const float*)d_in[3];
    const float* B2 = (const float*)d_in[4];
    float* out = (float*)d_out;

    const int H = in_sizes[2];
    const int D = in_sizes[4];
    const int Nrows = in_sizes[0] / D;
    const size_t szW1 = (size_t)in_sizes[1];
    const size_t szW2 = (size_t)in_sizes[3];
    const size_t szX  = (size_t)in_sizes[0];
    const size_t szX2 = (size_t)Nrows * H;

    char* ws = (char*)d_ws;
    size_t o = 0;
    auto carve = [&](size_t bytes) {
        size_t r = o;
        o += (bytes + 255) & ~(size_t)255;
        return r;
    };
    unsigned* wBits      = (unsigned*)(ws + carve(2 * sizeof(unsigned)));
    float* sX1           = (float*)(ws + carve((size_t)Nrows * sizeof(float)));
    unsigned* sX2bits    = (unsigned*)(ws + carve((size_t)Nrows * sizeof(unsigned)));
    signed char* W1q     = (signed char*)(ws + carve(szW1));
    signed char* W2q     = (signed char*)(ws + carve(szW2));
    signed char* X1q     = (signed char*)(ws + carve(szX));
    signed char* X2q     = (signed char*)(ws + carve(szX2));
    float* X2f           = (float*)(ws + carve(szX2 * sizeof(float)));
    (void)ws_size; (void)n_in; (void)out_size;

    init_zero_kernel<<<64, 256, 0, stream>>>(wBits, 2, sX2bits, Nrows);

    absmax_reduce_kernel<<<1024, 256, 0, stream>>>(W1, wBits + 0, szW1 >> 2);
    absmax_reduce_kernel<<<1024, 256, 0, stream>>>(W2, wBits + 1, szW2 >> 2);

    quant_global_kernel<<<4096, 256, 0, stream>>>(W1, W1q, wBits + 0, szW1 >> 2);
    quant_global_kernel<<<4096, 256, 0, stream>>>(W2, W2q, wBits + 1, szW2 >> 2);

    quant_rows_kernel<<<Nrows, 256, 0, stream>>>(x, X1q, sX1, D);

    {
        dim3 grid(H / 128, Nrows / 256);
        gemm_iu8_kernel<true><<<grid, 256, 0, stream>>>(X1q, W1q, sX1, wBits + 0, B1,
                                                        X2f, sX2bits, Nrows, H, D);
    }

    quant_rows_from_bits_kernel<<<8192, 256, 0, stream>>>(X2f, sX2bits, X2q, H, szX2 >> 2);

    {
        dim3 grid(D / 128, Nrows / 256);
        gemm_iu8_kernel<false><<<grid, 256, 0, stream>>>(X2q, W2q, (const float*)sX2bits,
                                                         wBits + 1, B2, out, nullptr,
                                                         Nrows, D, H);
    }
}
